// ConvAttention2D_86947317941212
// MI455X (gfx1250) — matrix operand order repaired and checked by reading, not compiled, not run on hardware
//
#include <hip/hip_runtime.h>
#include <hip/hip_bf16.h>
#include <stdint.h>

// ---------------- WMMA types & helpers (CDNA5 / gfx1250, wave32) ----------------
typedef __attribute__((ext_vector_type(16))) __bf16 v16b;
typedef __attribute__((ext_vector_type(8)))  float  v8f;

union Frag { v16b v; uint4 q[2]; };

#define WMMA_BF16(A, B, Cc) \
  __builtin_amdgcn_wmma_f32_16x16x32_bf16(false, (A), false, (B), (short)0, (Cc), false, false)

__device__ __forceinline__ uint16_t f2bf(float f) {
  uint32_t u = __float_as_uint(f);
  return (uint16_t)((u + 0x7fffu + ((u >> 16) & 1u)) >> 16);  // RNE
}
__device__ __forceinline__ v8f vzero() {
  v8f z = {0.f,0.f,0.f,0.f,0.f,0.f,0.f,0.f};
  return z;
}

// Problem constants
// B=4, C=128, T=1024, F=128, H=16, BH=64

// ---------------------------------------------------------------------------
// K0: pack conv weights into exact A-fragment layout (bf16 pairs per dword).
// conv1: [tap(9)][ck(4)][mt(3)][lane(32)][v(8)]  -> 27648 dwords
// conv2: [tap(9)][mt(8)][lane(32)][v(8)]         -> 18432 dwords
// A-fragment k order used here (B fragments match): row = lane&15, K = (v<<1) + e + ((lane>>4)<<4)
// ---------------------------------------------------------------------------
__global__ void pack_weights(const float* __restrict__ in_w, const float* __restrict__ out_w,
                             uint32_t* __restrict__ w1, uint32_t* __restrict__ w2) {
  int idx = blockIdx.x * 256 + threadIdx.x;
  if (idx < 27648) {
    int v = idx & 7, u = idx >> 3;
    int lane = u & 31; u >>= 5;
    int mt = u % 3; u /= 3;
    int ck = u & 3;  int tap = u >> 2;
    int hi = lane >> 4, row = (mt << 4) + (lane & 15);
    int ky = tap / 3, kx = tap % 3;
    int K0 = (v << 1) + (hi << 4);
    uint32_t pk = 0;
    for (int e = 0; e < 2; e++) {
      int c = (ck << 5) + K0 + e;
      float w = in_w[((row * 128 + c) * 3 + ky) * 3 + kx];
      pk |= ((uint32_t)f2bf(w)) << (16 * e);
    }
    w1[idx] = pk;
  } else if (idx < 27648 + 18432) {
    int i2 = idx - 27648;
    int v = i2 & 7, u = i2 >> 3;
    int lane = u & 31; u >>= 5;
    int mt = u & 7; int tap = u >> 3;
    int hi = lane >> 4, row = (mt << 4) + (lane & 15);
    int ky = tap / 3, kx = tap % 3;
    int K0 = (v << 1) + (hi << 4);
    uint32_t pk = 0;
    for (int e = 0; e < 2; e++) {
      int c = K0 + e;
      float w = out_w[((row * 32 + c) * 3 + ky) * 3 + kx];
      pk |= ((uint32_t)f2bf(w)) << (16 * e);
    }
    w2[i2] = pk;
  }
}

// ---------------------------------------------------------------------------
// K1: QKV 3x3 conv as implicit GEMM. One block = (b, t-row). M=48, N=128,
// K=1152 (9 taps x 4 ck-blocks of 32). X halo staged channel-major in LDS (bf16)
// so each B fragment is one contiguous 32B LDS read per lane.
// Emits relu'd (and q-scaled) q/k/v in row-major AND transposed bf16 layouts.
// ---------------------------------------------------------------------------
__global__ void __launch_bounds__(256) conv_qkv(const float* __restrict__ x,
    const float* __restrict__ in_b, const uint32_t* __restrict__ w1,
    uint16_t* __restrict__ qrm, uint16_t* __restrict__ krm, uint16_t* __restrict__ vrm,
    uint16_t* __restrict__ qt,  uint16_t* __restrict__ kt,  uint16_t* __restrict__ vt) {
  extern __shared__ __align__(16) char smem[];
  uint16_t* xs = (uint16_t*)smem;                 // [3][130][128] bf16, f padded +-1
  int tid = threadIdx.x;
  int b = blockIdx.x >> 10, t = blockIdx.x & 1023;
  const float* xb = x + (size_t)b * 128 * 1024 * 128;
  int f = tid & 127, c0 = tid >> 7;
  for (int r = 0; r < 3; r++) {
    int tt = t - 1 + r;
    if (tt >= 0 && tt < 1024) {
      for (int c = c0; c < 128; c += 2)
        xs[((r * 130) + f + 1) * 128 + c] = f2bf(xb[((size_t)c * 1024 + tt) * 128 + f]);
    } else {
      for (int c = c0; c < 128; c += 2)
        xs[((r * 130) + f + 1) * 128 + c] = 0;
    }
    if (tid < 128) { xs[(r * 130) * 128 + tid] = 0; xs[(r * 130 + 129) * 128 + tid] = 0; }
  }
  __syncthreads();

  int wave = tid >> 5, lane = tid & 31, hi = lane >> 4, ln = lane & 15;
  int fB = (wave << 4) + ln;                       // this wave's N column
  v8f acc[3]; acc[0] = vzero(); acc[1] = vzero(); acc[2] = vzero();
  for (int tap = 0; tap < 9; tap++) {
    int ky = tap / 3, dx = tap % 3 - 1;
    for (int ck = 0; ck < 4; ck++) {
      Frag fb;
      const uint16_t* bp = &xs[((ky * 130) + fB + dx + 1) * 128 + (ck << 5) + (hi << 4)];
      fb.q[0] = *(const uint4*)bp; fb.q[1] = *(const uint4*)(bp + 8);
      const uint32_t* wp = w1 + (size_t)((tap * 4 + ck) * 3) * 256 + lane * 8;
      for (int mt = 0; mt < 3; mt++) {
        Frag fa;
        fa.q[0] = *(const uint4*)(wp + mt * 256);
        fa.q[1] = *(const uint4*)(wp + mt * 256 + 4);
        acc[mt] = WMMA_BF16(fa.v, fb.v, acc[mt]);
      }
    }
  }
  const float scaling = 0.08838834764831843f;      // 128^-0.5 (relu(s*x)=s*relu(x))
  for (int mt = 0; mt < 3; mt++) {
    for (int r = 0; r < 8; r++) {
      int oc = (mt << 4) + r + (hi << 3);
      float val = fmaxf(acc[mt][r] + in_b[oc], 0.0f);
      int g = oc >> 4, h = oc & 15;
      if (g == 0) val *= scaling;
      size_t bh = (size_t)(b * 16 + h);
      uint16_t bv = f2bf(val);
      uint16_t* rm = (g == 0) ? qrm : ((g == 1) ? krm : vrm);
      uint16_t* tp = (g == 0) ? qt  : ((g == 1) ? kt  : vt);
      rm[(bh * 1024 + t) * 128 + fB] = bv;
      tp[(bh * 128 + fB) * 1024 + t] = bv;
    }
  }
}

// ---------------------------------------------------------------------------
// K2: time attention + averaged weights. One block = (b, 16-row t-tile),
// loops all 16 heads so aw_avg accumulates deterministically in LDS.
// Pass 1: flash-style row max/sum (per-wave over its 128 s-cols, then merged).
// Pass 2: recompute S, P=exp(S-m); accumulate normalized P into awsum; P@V via
// WMMA with P staged per-wave in LDS as A fragments; O reduced in fixed order.
// ---------------------------------------------------------------------------
__global__ void __launch_bounds__(256) attn_time(
    const uint16_t* __restrict__ qrm, const uint16_t* __restrict__ krm,
    const uint16_t* __restrict__ vt, uint16_t* __restrict__ attn,
    float* __restrict__ aw) {
  extern __shared__ __align__(16) char smem[];
  float*    awsum = (float*)smem;                        // 16*1024 f32
  float*    opart = awsum + 16 * 1024;                   // 8*16*128 f32
  uint16_t* pst   = (uint16_t*)(opart + 8 * 16 * 128);   // 8 waves x [16][40] bf16
  uint32_t* aq    = (uint32_t*)(pst + 8 * 16 * 40);      // 4*32*8 dwords (Q A-frags)
  float*    mW    = (float*)(aq + 1024);                 // 8*16
  float*    lW    = mW + 128;                            // 8*16
  float*    mF    = lW + 128;                            // 16
  float*    rF    = mF + 16;                             // 16

  int tid = threadIdx.x;
  int b = blockIdx.x >> 6, t0 = (blockIdx.x & 63) << 4;
  int wave = tid >> 5, lane = tid & 31, hi = lane >> 4, ln = lane & 15;
  for (int i = tid; i < 16 * 1024; i += 256) awsum[i] = 0.f;
  __syncthreads();

  for (int h = 0; h < 16; h++) {
    size_t bh = (size_t)(b * 16 + h);
    // stage Q tile as A fragments; zero per-wave O partials
    for (int i = tid; i < 1024; i += 256) {
      int v = i & 7, lu = (i >> 3) & 31, kc = i >> 8;
      int hiq = lu >> 4, rowq = lu & 15;
      int K0 = (v << 1) + (hiq << 4);
      aq[i] = *(const uint32_t*)&qrm[(bh * 1024 + t0 + rowq) * 128 + (kc << 5) + K0];
    }
    for (int i = tid; i < 8 * 16 * 128; i += 256) opart[i] = 0.f;
    __syncthreads();

    // ---- pass 1: per-wave online stats over its 128 s-columns ----
    float m[8], l[8];
    for (int r = 0; r < 8; r++) { m[r] = -3.0e38f; l[r] = 0.f; }
    int sBase = wave << 7;
    for (int sc = 0; sc < 8; sc++) {
      int s0 = sBase + (sc << 4);
      v8f S = vzero();
      for (int kc = 0; kc < 4; kc++) {
        Frag fa; const uint32_t* ap = aq + (kc << 8) + lane * 8;
        fa.q[0] = *(const uint4*)ap; fa.q[1] = *(const uint4*)(ap + 4);
        Frag fb; const uint16_t* bp = &krm[(bh * 1024 + s0 + ln) * 128 + (kc << 5) + (hi << 4)];
        fb.q[0] = *(const uint4*)bp; fb.q[1] = *(const uint4*)(bp + 8);
        S = WMMA_BF16(fa.v, fb.v, S);
      }
      for (int r = 0; r < 8; r++) {
        float xv = S[r];
        float mx = xv;
        mx = fmaxf(mx, __shfl_xor(mx, 1)); mx = fmaxf(mx, __shfl_xor(mx, 2));
        mx = fmaxf(mx, __shfl_xor(mx, 4)); mx = fmaxf(mx, __shfl_xor(mx, 8));
        float mn = fmaxf(m[r], mx);
        float p = __expf(xv - mn);
        float ps = p;
        ps += __shfl_xor(ps, 1); ps += __shfl_xor(ps, 2);
        ps += __shfl_xor(ps, 4); ps += __shfl_xor(ps, 8);
        l[r] = l[r] * __expf(m[r] - mn) + ps;
        m[r] = mn;
      }
    }
    if (ln == 0) {
      for (int r = 0; r < 8; r++) {
        int row = r + (hi << 3);
        mW[wave * 16 + row] = m[r];
        lW[wave * 16 + row] = l[r];
      }
    }
    __syncthreads();
    if (tid < 16) {
      float M = -3.0e38f;
      for (int w = 0; w < 8; w++) M = fmaxf(M, mW[w * 16 + tid]);
      float L = 0.f;
      for (int w = 0; w < 8; w++) L += lW[w * 16 + tid] * __expf(mW[w * 16 + tid] - M);
      mF[tid] = M; rF[tid] = 1.0f / L;
    }
    __syncthreads();
    float mrow[8], rrow[8];
    for (int r = 0; r < 8; r++) { mrow[r] = mF[r + (hi << 3)]; rrow[r] = rF[r + (hi << 3)]; }

    // ---- pass 2: recompute S, accumulate aw + O = P@V ----
    v8f O[8]; for (int nf = 0; nf < 8; nf++) O[nf] = vzero();
    int woff = wave * 16 * 40;
    for (int sc2 = 0; sc2 < 4; sc2++) {
      int s0 = sBase + (sc2 << 5);
      for (int half2 = 0; half2 < 2; half2++) {
        int s016 = s0 + (half2 << 4);
        v8f S = vzero();
        for (int kc = 0; kc < 4; kc++) {
          Frag fa; const uint32_t* ap = aq + (kc << 8) + lane * 8;
          fa.q[0] = *(const uint4*)ap; fa.q[1] = *(const uint4*)(ap + 4);
          Frag fb; const uint16_t* bp = &krm[(bh * 1024 + s016 + ln) * 128 + (kc << 5) + (hi << 4)];
          fb.q[0] = *(const uint4*)bp; fb.q[1] = *(const uint4*)(bp + 8);
          S = WMMA_BF16(fa.v, fb.v, S);
        }
        for (int r = 0; r < 8; r++) {
          int row = r + (hi << 3);
          float p = __expf(S[r] - mrow[r]);
          awsum[row * 1024 + s016 + ln] += p * rrow[r];          // normalized, per-wave cols
          pst[woff + row * 40 + (half2 << 4) + ln] = f2bf(p);    // unnormalized for P@V
        }
      }
      asm volatile("s_wait_dscnt 0" ::: "memory");               // LDS write -> cross-lane read
      Frag pa;
      const uint16_t* pb0 = &pst[woff + ln * 40 + (hi << 4)];
      pa.q[0] = *(const uint4*)pb0;
      pa.q[1] = *(const uint4*)(pb0 + 8);
      for (int nf = 0; nf < 8; nf++) {
        Frag fb; const uint16_t* vp = &vt[(bh * 128 + (nf << 4) + ln) * 1024 + s0 + (hi << 4)];
        fb.q[0] = *(const uint4*)vp; fb.q[1] = *(const uint4*)(vp + 8);
        O[nf] = WMMA_BF16(pa.v, fb.v, O[nf]);
      }
    }
    for (int nf = 0; nf < 8; nf++)
      for (int r = 0; r < 8; r++)
        opart[(wave * 16 + r + (hi << 3)) * 128 + (nf << 4) + ln] = O[nf][r];
    __syncthreads();
    // deterministic fixed-order reduction + normalize + store attn_t (channels 0..15)
    for (int i = tid; i < 2048; i += 256) {
      int row = i >> 7, fc = i & 127;
      float s = 0.f;
      for (int w = 0; w < 8; w++) s += opart[w * 2048 + row * 128 + fc];
      s *= rF[row];
      attn[(((size_t)(b * 32 + h)) * 1024 + t0 + row) * 128 + fc] = f2bf(s);
    }
    __syncthreads();
  }
  // averaged attention weights
  for (int i = tid; i < 16 * 1024; i += 256) {
    int row = i >> 10, s = i & 1023;
    aw[((size_t)b * 1024 + t0 + row) * 1024 + s] = awsum[i] * 0.0625f;
  }
}

// ---------------------------------------------------------------------------
// K3: frequency attention. One block = (b,h). Scores = Q^T K (128x128, K=1024),
// softmax rows in LDS, attn_f = AW @ V^T written transposed into channels 16..31.
// ---------------------------------------------------------------------------
__global__ void __launch_bounds__(256) attn_freq(
    const uint16_t* __restrict__ qt, const uint16_t* __restrict__ kt,
    const uint16_t* __restrict__ vrm, uint16_t* __restrict__ attn) {
  extern __shared__ __align__(16) char smem[];
  float*    Sf = (float*)smem;                    // [128][128] f32
  uint16_t* Pb = (uint16_t*)(Sf + 128 * 128);     // [128][136] bf16 (A-frag aligned)
  int tid = threadIdx.x;
  int b = blockIdx.x >> 4, h = blockIdx.x & 15;
  size_t bh = (size_t)(b * 16 + h);
  int wave = tid >> 5, lane = tid & 31, hi = lane >> 4, ln = lane & 15;

  v8f a[8]; for (int mt = 0; mt < 8; mt++) a[mt] = vzero();
  int f2 = (wave << 4) + ln;
  for (int kc = 0; kc < 32; kc++) {
    Frag fb; const uint16_t* bp = &kt[(bh * 128 + f2) * 1024 + (kc << 5) + (hi << 4)];
    fb.q[0] = *(const uint4*)bp; fb.q[1] = *(const uint4*)(bp + 8);
    for (int mt = 0; mt < 8; mt++) {
      Frag fa; const uint16_t* ap = &qt[(bh * 128 + (mt << 4) + ln) * 1024 + (kc << 5) + (hi << 4)];
      fa.q[0] = *(const uint4*)ap; fa.q[1] = *(const uint4*)(ap + 8);
      a[mt] = WMMA_BF16(fa.v, fb.v, a[mt]);
    }
  }
  for (int mt = 0; mt < 8; mt++)
    for (int r = 0; r < 8; r++)
      Sf[((mt << 4) + r + (hi << 3)) * 128 + f2] = a[mt][r];
  __syncthreads();
  if (tid < 128) {
    int row = tid;
    float M = -3.0e38f;
    for (int g = 0; g < 128; g++) M = fmaxf(M, Sf[row * 128 + g]);
    float L = 0.f;
    for (int g = 0; g < 128; g++) L += __expf(Sf[row * 128 + g] - M);
    float rinv = 1.f / L;
    for (int g = 0; g < 128; g++)
      Pb[row * 136 + g] = f2bf(__expf(Sf[row * 128 + g] - M) * rinv);
  }
  __syncthreads();
  for (int nt = 0; nt < 8; nt++) {
    int tcol = (wave << 7) + (nt << 4) + ln;
    v8f o[8]; for (int mt = 0; mt < 8; mt++) o[mt] = vzero();
    for (int kc = 0; kc < 4; kc++) {
      Frag fb; const uint16_t* vp = &vrm[(bh * 1024 + tcol) * 128 + (kc << 5) + (hi << 4)];
      fb.q[0] = *(const uint4*)vp; fb.q[1] = *(const uint4*)(vp + 8);
      for (int mt = 0; mt < 8; mt++) {
        Frag fa; const uint16_t* pp = &Pb[((mt << 4) + ln) * 136 + (kc << 5) + (hi << 4)];
        fa.q[0] = *(const uint4*)pp; fa.q[1] = *(const uint4*)(pp + 8);
        o[mt] = WMMA_BF16(fa.v, fb.v, o[mt]);
      }
    }
    for (int mt = 0; mt < 8; mt++)
      for (int r = 0; r < 8; r++) {
        int f1 = (mt << 4) + r + (hi << 3);
        attn[(((size_t)(b * 32 + 16 + h)) * 1024 + tcol) * 128 + f1] = f2bf(o[mt][r]);
      }
  }
}

// ---------------------------------------------------------------------------
// K4: output 3x3 conv (32 -> 128 ch) as implicit GEMM, relu, fp32 store.
// ---------------------------------------------------------------------------
__global__ void __launch_bounds__(256) conv_out(
    const uint16_t* __restrict__ attn, const float* __restrict__ out_b,
    const uint32_t* __restrict__ w2, float* __restrict__ out) {
  extern __shared__ __align__(16) char smem[];
  uint16_t* xs = (uint16_t*)smem;                 // [3][130][32] bf16
  int tid = threadIdx.x;
  int b = blockIdx.x >> 10, t = blockIdx.x & 1023;
  int f = tid & 127, c0 = tid >> 7;
  for (int r = 0; r < 3; r++) {
    int tt = t - 1 + r;
    for (int c = c0; c < 32; c += 2) {
      uint16_t vx = 0;
      if (tt >= 0 && tt < 1024)
        vx = attn[(((size_t)(b * 32 + c)) * 1024 + tt) * 128 + f];
      xs[((r * 130) + f + 1) * 32 + c] = vx;
    }
    if (tid < 32) { xs[(r * 130) * 32 + tid] = 0; xs[(r * 130 + 129) * 32 + tid] = 0; }
  }
  __syncthreads();
  int wave = tid >> 5, lane = tid & 31, hi = lane >> 4, ln = lane & 15;
  int fB = (wave << 4) + ln;
  v8f acc[8]; for (int mt = 0; mt < 8; mt++) acc[mt] = vzero();
  for (int tap = 0; tap < 9; tap++) {
    int ky = tap / 3, dx = tap % 3 - 1;
    Frag fb; const uint16_t* bp = &xs[((ky * 130) + fB + dx + 1) * 32 + (hi << 4)];
    fb.q[0] = *(const uint4*)bp; fb.q[1] = *(const uint4*)(bp + 8);
    const uint32_t* wp = w2 + (size_t)(tap * 8) * 256 + lane * 8;
    for (int mt = 0; mt < 8; mt++) {
      Frag fa;
      fa.q[0] = *(const uint4*)(wp + mt * 256);
      fa.q[1] = *(const uint4*)(wp + mt * 256 + 4);
      acc[mt] = WMMA_BF16(fa.v, fb.v, acc[mt]);
    }
  }
  for (int mt = 0; mt < 8; mt++)
    for (int r = 0; r < 8; r++) {
      int oc = (mt << 4) + r + (hi << 3);
      float val = fmaxf(acc[mt][r] + out_b[oc], 0.f);
      out[(((size_t)(b * 128 + oc)) * 1024 + t) * 128 + fB] = val;
    }
}

// ---------------------------------------------------------------------------
extern "C" void kernel_launch(void* const* d_in, const int* in_sizes, int n_in,
                              void* d_out, int out_size, void* d_ws, size_t ws_size,
                              hipStream_t stream) {
  const float* query = (const float*)d_in[0];
  const float* in_w  = (const float*)d_in[1];
  const float* in_b  = (const float*)d_in[2];
  const float* out_w = (const float*)d_in[3];
  const float* out_b = (const float*)d_in[4];
  float* out = (float*)d_out;
  float* aw  = out + (size_t)4 * 128 * 1024 * 128;   // aw_avg after main output

  char* ws = (char*)d_ws;
  const size_t SEG = 16777216ull;                    // 16MB per bf16 qkv buffer
  uint16_t* qrm  = (uint16_t*)(ws + 0 * SEG);
  uint16_t* krm  = (uint16_t*)(ws + 1 * SEG);
  uint16_t* vrm  = (uint16_t*)(ws + 2 * SEG);
  uint16_t* qt   = (uint16_t*)(ws + 3 * SEG);
  uint16_t* kt   = (uint16_t*)(ws + 4 * SEG);
  uint16_t* vt   = (uint16_t*)(ws + 5 * SEG);
  uint16_t* attn = (uint16_t*)(ws + 6 * SEG);        // 32MB: [4][32][1024][128] bf16
  uint32_t* w1   = (uint32_t*)(ws + 8 * SEG);        // 110592 B
  uint32_t* w2   = (uint32_t*)(ws + 8 * SEG + 131072);

  pack_weights<<<dim3(180), dim3(256), 0, stream>>>(in_w, out_w, w1, w2);
  conv_qkv   <<<dim3(4096), dim3(256), 3 * 130 * 128 * 2, stream>>>(
      query, in_b, w1, qrm, krm, vrm, qt, kt, vt);
  attn_time  <<<dim3(256), dim3(256), 146560, stream>>>(qrm, krm, vt, attn, aw);
  attn_freq  <<<dim3(64),  dim3(256), 100352, stream>>>(qt, kt, vrm, attn);
  conv_out   <<<dim3(4096), dim3(256), 3 * 130 * 32 * 2, stream>>>(attn, out_b, w2, out);
}
